// Int8Linear_16724602650923
// MI455X (gfx1250) — matrix operand order repaired and checked by reading, not compiled, not run on hardware
//
#include <hip/hip_runtime.h>
#include <stdint.h>

// y[m,n] = (sum_k x[m,k] * W[n,k]) * scale[n],  fp32 accum, fp16 out
// M = B*S = 4096, K = 4096, N = 11008
// Compute-bound (3.7e11 FLOP vs ~170MB traffic): fp16 WMMA path, int8 weights
// dequantized exactly in registers with a packed magic-bias trick, per-channel
// scale applied after the K reduction (matches reference numerics).

typedef __attribute__((ext_vector_type(16))) _Float16 v16h;
typedef __attribute__((ext_vector_type(8)))  _Float16 v8h;
typedef __attribute__((ext_vector_type(2)))  _Float16 v2h;
typedef __attribute__((ext_vector_type(8)))  float    v8f;
typedef __attribute__((ext_vector_type(4)))  uint32_t v4u;

#define K_DIM 4096
#define N_DIM 11008

// Each wave owns a 64x64 output tile: 4 (M) x 4 (N) WMMA fragments.
// Block = 128 threads = 4 waves arranged 2x2 -> 128x128 block tile.
__global__ __launch_bounds__(128) void int8linear_wmma_kernel(
    const _Float16* __restrict__ X,   // [M, K] fp16 row-major
    const int8_t*   __restrict__ W,   // [N, K] int8 row-major
    const float*    __restrict__ S,   // [N] fp32
    _Float16*       __restrict__ Y)   // [M, N] fp16 row-major
{
    const uint32_t lane     = threadIdx.x & 31u;
    const uint32_t wave     = threadIdx.x >> 5;   // 0..3
    const uint32_t wm       = wave >> 1;          // 0..1
    const uint32_t wn       = wave & 1u;          // 0..1
    const uint32_t half_sel = lane >> 4;          // 0: lanes 0-15, 1: lanes 16-31
    const uint32_t lr       = lane & 15u;

    const uint32_t m0 = blockIdx.y * 128u + wm * 64u;
    const uint32_t n0 = blockIdx.x * 128u + wn * 64u;

    // 32-bit BYTE offsets from the uniform bases (all tensors < 4GB):
    // steers codegen to global_load saddr+voffset instead of flat_load.
    uint32_t offA[4], offB[4];
#pragma unroll
    for (int i = 0; i < 4; ++i)
        offA[i] = ((m0 + 16u * (uint32_t)i + lr) * (uint32_t)K_DIM + half_sel * 16u) * 2u;
#pragma unroll
    for (int j = 0; j < 4; ++j)
        offB[j] = (n0 + 16u * (uint32_t)j + lr) * (uint32_t)K_DIM + half_sel * 16u;

    v8f acc[4][4];
    const v8f zero = {};
#pragma unroll
    for (int i = 0; i < 4; ++i)
#pragma unroll
        for (int j = 0; j < 4; ++j)
            acc[i][j] = zero;

    // magic-bias dequant constant: (1024+u) - 1152 = u - 128 = signed int8, exact
    const v2h bias = { (_Float16)(-1152.0f), (_Float16)(-1152.0f) };

    for (uint32_t k = 0; k < (uint32_t)K_DIM; k += 32u) {
        if (k + 32u < (uint32_t)K_DIM) {
            __builtin_prefetch((const char*)X + (offA[0] + (k + 32u) * 2u), 0, 3);
            __builtin_prefetch((const char*)W + (offB[0] + (k + 32u)), 0, 3);
        }

        // ---- A fragments (lane = row lr; lo lanes K{0..15},
        //      hi lanes K{16..31}; two contiguous 16B chunks 16B apart) ----
        v16h a[4];
#pragma unroll
        for (int i = 0; i < 4; ++i) {
            const char* p = (const char*)X + (offA[i] + k * 2u);
            v8h lo = *(const v8h*)p;
            v8h hi = *(const v8h*)(p + 16);
            a[i] = __builtin_shufflevector(lo, hi,
                                           0, 1, 2, 3, 4, 5, 6, 7,
                                           8, 9, 10, 11, 12, 13, 14, 15);
        }

        // ---- B fragments: 16 contiguous int8 (column n of B = row n of W),
        //      packed exact dequant: xor 0x80, v_perm with 0x64 high bytes,
        //      v_pk_add_f16 of -1152.0  => 5 VALU per 4 weights ----
        v16h b[4];
#pragma unroll
        for (int j = 0; j < 4; ++j) {
            v4u q = *(const v4u*)((const char*)W + (offB[j] + k));
            union { v16h h; uint32_t u[8]; } t;
#pragma unroll
            for (int d = 0; d < 4; ++d) {
                uint32_t wv  = q[d] ^ 0x80808080u;                      // bias to [0,255]
                // {0x64,b1,0x64,b0} and {0x64,b3,0x64,b2}: halves = 0x6400 + u
                uint32_t plo = __builtin_amdgcn_perm(0x64646464u, wv, 0x04010400u);
                uint32_t phi = __builtin_amdgcn_perm(0x64646464u, wv, 0x04030402u);
                v2h hlo = __builtin_bit_cast(v2h, plo) + bias;          // v_pk_add_f16
                v2h hhi = __builtin_bit_cast(v2h, phi) + bias;
                t.u[2 * d + 0] = __builtin_bit_cast(uint32_t, hlo);     // K=4d,4d+1
                t.u[2 * d + 1] = __builtin_bit_cast(uint32_t, hhi);     // K=4d+2,4d+3
            }
            b[j] = t.h;
        }

        // ---- 16 WMMAs: D = A x B + C, fp32 accumulation ----
#pragma unroll
        for (int i = 0; i < 4; ++i)
#pragma unroll
            for (int j = 0; j < 4; ++j)
                acc[i][j] = __builtin_amdgcn_wmma_f32_16x16x32_f16(
                    /*neg_a=*/false, a[i],
                    /*neg_b=*/false, b[j],
                    /*c_mod=*/(short)0, acc[i][j],
                    /*reuse_a=*/false, /*reuse_b=*/false);
    }

    // ---- Epilogue: acc * scale[n] -> fp16 store ----
    // C/D layout: VGPR v, lanes 0-15: (M=v, N=lane); lanes 16-31: (M=8+v, N=lane-16).
#pragma unroll
    for (int j = 0; j < 4; ++j) {
        const uint32_t col = n0 + 16u * (uint32_t)j + lr;
        const float s = S[col];
#pragma unroll
        for (int i = 0; i < 4; ++i) {
            const uint32_t row_base = m0 + 16u * (uint32_t)i + half_sel * 8u;
#pragma unroll
            for (int v = 0; v < 8; ++v) {
                const uint32_t offY =
                    ((row_base + (uint32_t)v) * (uint32_t)N_DIM + col) * 2u;
                *(_Float16*)((char*)Y + offY) = (_Float16)(acc[i][j][v] * s);
            }
        }
    }
}

extern "C" void kernel_launch(void* const* d_in, const int* in_sizes, int n_in,
                              void* d_out, int out_size, void* d_ws, size_t ws_size,
                              hipStream_t stream) {
    const _Float16* x = (const _Float16*)d_in[0];  // fp16 [B,S,K] flat = [M,K]
    const int8_t*   w = (const int8_t*)d_in[1];    // int8 [N,K]
    const float*    s = (const float*)d_in[2];     // fp32 [N]
    _Float16*       y = (_Float16*)d_out;          // fp16 [M,N]

    const int M = in_sizes[0] / K_DIM;             // 4096 for the reference shapes

    dim3 block(128);                               // 4 waves (wave32)
    dim3 grid(N_DIM / 128, M / 128);               // (86, 32), exact tiling
    int8linear_wmma_kernel<<<grid, block, 0, stream>>>(x, w, s, y);
}
